// SOFIN_33474975105109
// MI455X (gfx1250) — matrix operand order repaired and checked by reading, not compiled, not run on hardware
//
#include <hip/hip_runtime.h>

typedef __attribute__((ext_vector_type(16))) _Float16 v16h;
typedef __attribute__((ext_vector_type(8)))  float    v8f;
typedef __attribute__((ext_vector_type(4)))  int      v4i;

// workspace layout (bytes)
#define WH_OFF    0         // 512*128 f16 tiled B-operand for consequent, 131072 B
#define BANT_OFF  131072    // 256*16 f16 tiled B-operand for antecedent, 8192 B
#define CONST_OFF 139264    // 16 f32 per-rule constant sum(a*c^2)

#if defined(__has_builtin)
#if __has_builtin(__builtin_amdgcn_global_load_async_to_lds_b128) && \
    __has_builtin(__builtin_amdgcn_s_wait_asynccnt)
#define SOFIN_ASYNC 1
#endif
#endif

// ---------------------------------------------------------------------------
// Prep: build f16 B-operands in WMMA register order + antecedent constants.
// B-operand layout used here (contiguous k order, A loaded the same way):
//   lane holds column n = lane&15 ; element e holds k = kt*32 + 16*(lane>>4) + e
//   flat f16 index = ((nt*4 + kt)*32 + lane)*16 + e   -> 32 B contiguous per lane
// ---------------------------------------------------------------------------
__global__ void sofin_prep(const float* __restrict__ centers,
                           const float* __restrict__ widths,
                           const float* __restrict__ W,
                           unsigned char* __restrict__ ws) {
  _Float16* wh   = (_Float16*)(ws + WH_OFF);
  _Float16* bant = (_Float16*)(ws + BANT_OFF);
  float*    cst  = (float*)(ws + CONST_OFF);
  int idx = blockIdx.x * 256 + threadIdx.x;
  if (idx < 65536) {
    // consequent weights: column n = r*32+o, k = d
    int e = idx & 15, lane = (idx >> 4) & 31, kt = (idx >> 9) & 3, nt = idx >> 11;
    int n = nt * 16 + (lane & 15);
    int r = n >> 5, o = n & 31;
    int k = kt * 32 + ((lane >> 4) << 4) + e;
    wh[idx] = (_Float16)W[(r * 32 + o) * 128 + k];
  } else if (idx < 69632) {
    // antecedent B: rows 0..127 -> a = 0.5/w^2 (multiplies s^2),
    //               rows 128..255 -> -2*a*c   (multiplies s)
    int t = idx - 65536;
    int e = t & 15, lane = (t >> 4) & 31, kt = t >> 9;   // kt 0..7 (K=256)
    int r = lane & 15;
    int k = kt * 32 + ((lane >> 4) << 4) + e;
    int d = (k < 128) ? k : (k - 128);
    float w = widths[r * 128 + d];
    float a = 0.5f / (w * w);
    float v = (k < 128) ? a : (-2.0f * a * centers[r * 128 + d]);
    bant[t] = (_Float16)v;
  } else if (idx < 69648) {
    int r = idx - 69632;
    float acc = 0.f;
    for (int d = 0; d < 128; ++d) {
      float w = widths[r * 128 + d], c = centers[r * 128 + d];
      acc += (0.5f / (w * w)) * c * c;
    }
    cst[r] = acc;
  }
}

// ---------------------------------------------------------------------------
// Main: 8 waves, 128 batch rows per block. Wh staged in LDS (async when
// available) so the consequent B-tiles are served by LDS, not L2.
// ---------------------------------------------------------------------------
__global__ __launch_bounds__(256)
void sofin_main(const float* __restrict__ x,
                const float* __restrict__ bias,
                const unsigned char* __restrict__ ws,
                float* __restrict__ out, int nrows) {
  // padded rows: 136/264 halves -> 272/528 B row stride (16B aligned, bank-spread)
  __shared__ __align__(16) _Float16 lds_xh[128 * 136];   //  34816 B
  __shared__ __align__(16) _Float16 lds_ant[128 * 264];  //  67584 B
  __shared__ __align__(16) _Float16 lds_wh[512 * 128];   // 131072 B

  const float* cst   = (const float*)(ws + CONST_OFF);
  const uint4* bantq = (const uint4*)(ws + BANT_OFF);

  const int tid = threadIdx.x;
  const int lane = tid & 31;
  const int wave = tid >> 5;
  const int rowBase = blockIdx.x * 128;

  // ---- Kick off Wh (global -> LDS) copy: 8192 x b128 ----
#ifdef SOFIN_ASYNC
  {
    typedef __attribute__((address_space(1))) v4i gv4i;
    typedef __attribute__((address_space(3))) v4i lv4i;
    gv4i* g = (gv4i*)(ws + WH_OFF);
    lv4i* l = (lv4i*)lds_wh;
#pragma unroll 4
    for (int i = 0; i < 32; ++i) {
      int t = tid + i * 256;
      __builtin_amdgcn_global_load_async_to_lds_b128(g + t, l + t, 0, 0);
    }
  }
#else
  {
    const uint4* gq = (const uint4*)(ws + WH_OFF);
    uint4* lq = (uint4*)lds_wh;
#pragma unroll 4
    for (int i = 0; i < 32; ++i) {
      int t = tid + i * 256;
      lq[t] = gq[t];
    }
  }
#endif

  // ---- Phase 1: stage x tile, compute sigmoid once ----
  const float4* x4 = (const float4*)x;
  for (int i = 0; i < 16; ++i) {
    int idx = tid + i * 256;          // 0..4095 float4s
    int rl  = idx >> 5;               // local row 0..127
    int c4  = (idx & 31) << 2;        // column 0..124 step 4
    int grow = rowBase + rl; if (grow >= nrows) grow = nrows - 1;
    float4 v = x4[grow * 32 + (c4 >> 2)];
    float vs[4] = {v.x, v.y, v.z, v.w};
    _Float16* px  = &lds_xh[rl * 136 + c4];
    _Float16* ps2 = &lds_ant[rl * 264 + c4];
    _Float16* ps  = &lds_ant[rl * 264 + 128 + c4];
#pragma unroll
    for (int c = 0; c < 4; ++c) {
      float xv = vs[c];
      float s  = 1.0f / (1.0f + __expf(-xv));
      px[c]  = (_Float16)xv;
      ps2[c] = (_Float16)(s * s);
      ps[c]  = (_Float16)s;
    }
  }
  __syncthreads();

  union AV { v16h v; uint4 u[2]; };
  const int mrow  = (wave << 4) + (lane & 15);
  const int khalf = (lane >> 4) << 4;          // A layout: lane half -> K +16

  // ---- Antecedent GEMM: [s^2, s] (16x256) x Bant (256x16) ----
  v8f facc = {};
  for (int kt = 0; kt < 8; ++kt) {
    AV a;
    const _Float16* ap = &lds_ant[mrow * 264 + kt * 32 + khalf];
    a.u[0] = *(const uint4*)ap;          // k .. k+7
    a.u[1] = *(const uint4*)(ap + 8);    // k+8 .. k+15
    AV bm;
    bm.u[0] = bantq[(kt * 32 + lane) * 2];
    bm.u[1] = bantq[(kt * 32 + lane) * 2 + 1];
    facc = __builtin_amdgcn_wmma_f32_16x16x32_f16(false, a.v, false, bm.v,
                                                  (short)0, facc, false, false);
  }
  // fs = -(gemm + sum(a c^2)) / D ; softmax across the 16 rule lanes
  float ca = cst[lane & 15];
  float fsw[8];
#pragma unroll
  for (int j = 0; j < 8; ++j) {
    float fv = -(facc[j] + ca) * (1.0f / 128.0f);
    float m = fv;
#pragma unroll
    for (int msk = 1; msk < 16; msk <<= 1)
      m = fmaxf(m, __shfl_xor(m, msk, 32));
    float e = __expf(fv - m);
    float s = e;
#pragma unroll
    for (int msk = 1; msk < 16; msk <<= 1)
      s += __shfl_xor(s, msk, 32);
    fsw[j] = e / s;
  }

  // ---- Wh copy must be complete & visible before the consequent GEMM ----
#ifdef SOFIN_ASYNC
  __builtin_amdgcn_s_wait_asynccnt(0);
#endif
  __syncthreads();

  // ---- Consequent GEMM: A tiles resident in VGPRs, B tiles from LDS ----
  AV a4[4];
#pragma unroll
  for (int kt = 0; kt < 4; ++kt) {
    const _Float16* ap = &lds_xh[mrow * 136 + kt * 32 + khalf];
    a4[kt].u[0] = *(const uint4*)ap;
    a4[kt].u[1] = *(const uint4*)(ap + 8);
  }

  const uint4* whq = (const uint4*)lds_wh;
  float oa[2][8] = {};
  for (int nt = 0; nt < 32; ++nt) {
    int r = nt >> 1, oh = nt & 1;                      // rule, O-half
    float bv = bias[nt * 16 + (lane & 15)];            // == b[r][oh*16 + n]
    v8f c = {bv, bv, bv, bv, bv, bv, bv, bv};          // bias pre-loaded into C
#pragma unroll
    for (int kt = 0; kt < 4; ++kt) {
      AV bm;
      int t = ((nt * 4 + kt) * 32 + lane) * 2;
      bm.u[0] = whq[t];
      bm.u[1] = whq[t + 1];
      c = __builtin_amdgcn_wmma_f32_16x16x32_f16(false, a4[kt].v, false, bm.v,
                                                 (short)0, c, false, false);
    }
    int srcl = (lane & 16) | r;                        // lane holding fsw for rule r, same M-half
#pragma unroll
    for (int j = 0; j < 8; ++j) {
      float cq  = 1.0f / (1.0f + __expf(-c[j]));
      float wgt = __shfl(fsw[j], srcl, 32);
      oa[oh][j] += wgt * cq;
    }
  }

  // ---- final sigmoid + store ----
#pragma unroll
  for (int j = 0; j < 8; ++j) {
    int row = rowBase + (wave << 4) + j + ((lane >> 4) << 3);
    if (row < nrows) {
      out[row * 32 + (lane & 15)]      = 1.0f / (1.0f + __expf(-oa[0][j]));
      out[row * 32 + 16 + (lane & 15)] = 1.0f / (1.0f + __expf(-oa[1][j]));
    }
  }
}

extern "C" void kernel_launch(void* const* d_in, const int* in_sizes, int n_in,
                              void* d_out, int out_size, void* d_ws, size_t ws_size,
                              hipStream_t stream) {
  const float* x       = (const float*)d_in[0];
  const float* centers = (const float*)d_in[1];
  const float* widths  = (const float*)d_in[2];
  const float* W       = (const float*)d_in[3];
  const float* b       = (const float*)d_in[4];
  float* out = (float*)d_out;
  unsigned char* ws = (unsigned char*)d_ws;
  int nrows = in_sizes[0] / 128;

  sofin_prep<<<273, 256, 0, stream>>>(centers, widths, W, ws);   // 273*256 >= 69648
  int blocks = (nrows + 127) / 128;
  sofin_main<<<blocks, 256, 0, stream>>>(x, b, ws, out, nrows);
}
